// DGCNN_GraphLayer_90082644066706
// MI455X (gfx1250) — matrix operand order repaired and checked by reading, not compiled, not run on hardware
//
#include <hip/hip_runtime.h>
#include <hip/hip_bf16.h>

// ---------------------------------------------------------------------------
// DGCNN on gfx1250 (MI455X): all GEMMs via v_wmma_f32_16x16x32_f16 (wave32).
// Activations stored (N, C) channel-contiguous so every WMMA fragment is
// two contiguous 16-byte vector loads (global_load_b128) instead of 16
// scalar d16 loads.
// ---------------------------------------------------------------------------

typedef __attribute__((ext_vector_type(16))) _Float16 v16h;
typedef __attribute__((ext_vector_type(8)))  _Float16 v8h;
typedef __attribute__((ext_vector_type(8)))  float    v8f;

#define BB 4
#define NN 2048
#define KK 20

__device__ __forceinline__ v8f wmma16(v16h a, v16h b, v8f c) {
  // D = A(16x32 f16) * B(32x16 f16) + C(16x16 f32)
  return __builtin_amdgcn_wmma_f32_16x16x32_f16(false, a, false, b, (short)0, c, false, false);
}

__device__ __forceinline__ v16h combine(v8h lo, v8h hi) {
  return __builtin_shufflevector(lo, hi, 0, 1, 2, 3, 4, 5, 6, 7, 8, 9, 10, 11, 12, 13, 14, 15);
}

// A-fragment (16x32 f16): lane row pointer `row`, chunk base cb, half `hi`.
// Lane loads K = cb + hi*16 + [0..15]  (contiguous; B uses the same order).
__device__ __forceinline__ v16h load_afrag(const _Float16* __restrict__ row, int cb, int hi) {
  v8h lo = *(const v8h*)(row + cb + hi * 16);
  v8h hh = *(const v8h*)(row + cb + hi * 16 + 8);
  return combine(lo, hh);
}

// B-fragment (32x16 f16): lane column's row pointer, K = cb + hi*16 + [0..15].
__device__ __forceinline__ v16h load_bfrag(const _Float16* __restrict__ row, int cb, int hi) {
  const _Float16* p = row + cb + hi * 16;
  return combine(*(const v8h*)p, *(const v8h*)(p + 8));
}

// ---------------------------------------------------------------------------
// Prep kernels
// ---------------------------------------------------------------------------

// x (B,3,N) f32 -> xpad (B,N,32) f16, channels 3..31 zero
__global__ void prep_x_kernel(const float* __restrict__ x, _Float16* __restrict__ xp) {
  int t = blockIdx.x * blockDim.x + threadIdx.x;       // B*N threads
  int b = t / NN, n = t - b * NN;
  _Float16* row = xp + (long)t * 32;
  for (int c = 0; c < 32; ++c) {
    float v = (c < 3) ? x[((long)b * 3 + c) * NN + n] : 0.0f;
    row[c] = (_Float16)v;
  }
}

// W (co, 2*Creal) f32 -> Wp (co, twoCpad) f16 row-major, [central | diff] at
// column offsets 0 and Cinpad, zeros elsewhere.
__global__ void prep_w_kernel(const float* __restrict__ W, _Float16* __restrict__ Wp,
                              int co, int Creal, int Cinpad, int twoCpad) {
  int t = blockIdx.x * blockDim.x + threadIdx.x;
  if (t >= co * twoCpad) return;
  int o = t / twoCpad, c = t - o * twoCpad;
  float v = 0.0f;
  if (c < Creal)                              v = W[o * 2 * Creal + c];
  else if (c >= Cinpad && c < Cinpad + Creal) v = W[o * 2 * Creal + Creal + (c - Cinpad)];
  Wp[t] = (_Float16)v;
}

__global__ void w_flat_kernel(const float* __restrict__ W, _Float16* __restrict__ Wp, long n) {
  long t = (long)blockIdx.x * blockDim.x + threadIdx.x;
  if (t < n) Wp[t] = (_Float16)W[t];
}

// sq[b*N+n] = sum_c X[n,c]^2   (X slice: row stride RS halves, Cin channels)
__global__ void sqnorm_kernel(const _Float16* __restrict__ X, int RS,
                              float* __restrict__ sq, int C) {
  int t = blockIdx.x * blockDim.x + threadIdx.x;       // B*N threads
  const _Float16* row = X + (long)t * RS;
  float s = 0.0f;
  for (int c = 0; c < C; ++c) { float v = (float)row[c]; s += v * v; }
  sq[t] = s;
}

// ---------------------------------------------------------------------------
// Pairwise distance:  dist[n,m] = sq[n] + sq[m] - 2 * (X X^T)[n,m]   (WMMA)
// X layout (B*N, RS) rows; grid (N/128, N/16, B), block 256 (8 waves).
// ---------------------------------------------------------------------------
__global__ void gram_dist_kernel(const _Float16* __restrict__ X, int RS,
                                 const float* __restrict__ sq,
                                 float* __restrict__ dist, int C) {
  int wave = threadIdx.x >> 5, lane = threadIdx.x & 31;
  int l15 = lane & 15, hi = lane >> 4;
  int b  = blockIdx.z;
  int n0 = (blockIdx.x * 8 + wave) * 16;
  int m0 = blockIdx.y * 16;
  const _Float16* rowN = X + ((long)b * NN + n0 + l15) * RS;
  const _Float16* rowM = X + ((long)b * NN + m0 + l15) * RS;

  v8f acc = {};
  for (int cb = 0; cb < C; cb += 32) {
    v16h a  = load_afrag(rowN, cb, hi);
    v16h bm = load_bfrag(rowM, cb, hi);
    acc = wmma16(a, bm, acc);
  }

  const float* sqb = sq + b * NN;
  float sqm = sqb[m0 + l15];
#pragma unroll
  for (int r = 0; r < 8; ++r) {
    int n = n0 + r + (hi << 3);
    dist[((long)b * NN + n) * NN + (m0 + l15)] = sqb[n] + sqm - 2.0f * acc[r];
  }
}

// ---------------------------------------------------------------------------
// Top-K (k=20 smallest dist) per row; one wave per (b,n).
// ---------------------------------------------------------------------------
__global__ void topk_kernel(const float* __restrict__ dist, int* __restrict__ idx) {
  int wid  = (blockIdx.x * blockDim.x + threadIdx.x) >> 5;   // B*N waves
  int lane = threadIdx.x & 31;
  int b = wid / NN, n = wid - b * NN;
  const float* row = dist + ((long)b * NN + n) * NN;

  float v[KK]; int vi[KK];
#pragma unroll
  for (int i = 0; i < KK; ++i) { v[i] = 3.4e38f; vi[i] = 0x7fffffff; }
  float curmax = 3.4e38f; int maxpos = 0;

  for (int m = lane; m < NN; m += 32) {
    if (m + 1024 < NN) __builtin_prefetch(row + m + 1024, 0, 1);
    float d = row[m];
    if (d < curmax) {
      v[maxpos] = d; vi[maxpos] = m;
      curmax = v[0]; maxpos = 0;
      for (int i = 1; i < KK; ++i) if (v[i] > curmax) { curmax = v[i]; maxpos = i; }
    }
  }
  // insertion sort ascending
  for (int i = 1; i < KK; ++i) {
    float key = v[i]; int ki = vi[i]; int j = i - 1;
    while (j >= 0 && v[j] > key) { v[j + 1] = v[j]; vi[j + 1] = vi[j]; --j; }
    v[j + 1] = key; vi[j + 1] = ki;
  }
  // 20 rounds of wave-wide arg-min merge
  int head = 0;
  int* out = idx + ((long)b * NN + n) * KK;
  for (int kk = 0; kk < KK; ++kk) {
    float hv = (head < KK) ? v[head] : 3.4e38f;
    int   hm = (head < KK) ? vi[head] : 0x7fffffff;
    float bv = hv; int bm = hm;
#pragma unroll
    for (int off = 16; off > 0; off >>= 1) {
      float ov = __shfl_xor(bv, off, 32);
      int   om = __shfl_xor(bm, off, 32);
      if (ov < bv || (ov == bv && om < bm)) { bv = ov; bm = om; }
    }
    if (hm == bm) head++;          // winner (unique m) advances
    if (lane == 0) out[kk] = bm;
  }
}

// ---------------------------------------------------------------------------
// Edge conv (templated on NCH = twoCpad/32; Cin = NCH*16):
//   y[o, j] = sum_c Wp[o,c] * e[c,j],  j = n*20 + kk  (per batch)
//   e[c,j]: c<Cin -> X[n,c] ; else X[m,c-Cin] - X[n,c-Cin]
// W fragments preloaded per wave (invariant over column tiles).
// grid (Co/16, strips, B), block 256; each wave handles TPW col-tiles.
// ---------------------------------------------------------------------------
template <int NCH>
__device__ __forceinline__ v16h edge_bfrag(const _Float16* __restrict__ Xn,
                                           const _Float16* __restrict__ Xm,
                                           int k, int hi) {
  const int Cin = NCH * 16;
  int c = k * 32 + hi * 16;
  if (c < Cin) {                      // wave-uniform for even NCH
    return combine(*(const v8h*)(Xn + c), *(const v8h*)(Xn + c + 8));
  } else {
    int c2 = c - Cin;
    v8h d0 = *(const v8h*)(Xm + c2)     - *(const v8h*)(Xn + c2);
    v8h d1 = *(const v8h*)(Xm + c2 + 8) - *(const v8h*)(Xn + c2 + 8);
    return combine(d0, d1);
  }
}

template <int NCH>
__global__ void edge_stats_kernel(const _Float16* __restrict__ X, int RS,
                                  const int* __restrict__ idx,
                                  const _Float16* __restrict__ Wp,
                                  float* __restrict__ sum, float* __restrict__ sumsq,
                                  int TPW) {
  const int twoC = NCH * 32;
  int wave = threadIdx.x >> 5, lane = threadIdx.x & 31;
  int l15 = lane & 15, hi = lane >> 4;
  int b = blockIdx.z, obase = blockIdx.x * 16;
  const _Float16* Xb = X + (long)b * NN * RS;
  const int* idxb = idx + (long)b * NN * KK;
  const int CT = NN * KK / 16;

  v16h af[NCH];
  const _Float16* wrow = Wp + (long)(obase + l15) * twoC;
#pragma unroll
  for (int k = 0; k < NCH; ++k) af[k] = load_afrag(wrow, k * 32, hi);

  float s[8] = {0, 0, 0, 0, 0, 0, 0, 0}, q[8] = {0, 0, 0, 0, 0, 0, 0, 0};
  for (int it = 0; it < TPW; ++it) {
    int t = (blockIdx.y * 8 + wave) * TPW + it;
    if (t >= CT) break;
    int j = t * 16 + l15;
    int n = j / KK;
    int m = idxb[j];
    const _Float16* Xn = Xb + (long)n * RS;
    const _Float16* Xm = Xb + (long)m * RS;
    v8f acc = {};
#pragma unroll
    for (int k = 0; k < NCH; ++k)
      acc = wmma16(af[k], edge_bfrag<NCH>(Xn, Xm, k, hi), acc);
#pragma unroll
    for (int r = 0; r < 8; ++r) { float y = acc[r]; s[r] += y; q[r] += y * y; }
  }
#pragma unroll
  for (int r = 0; r < 8; ++r)
#pragma unroll
    for (int off = 8; off > 0; off >>= 1) {
      s[r] += __shfl_xor(s[r], off, 16);
      q[r] += __shfl_xor(q[r], off, 16);
    }
  if (l15 == 0) {
#pragma unroll
    for (int r = 0; r < 8; ++r) {
      int o = obase + r + (hi << 3);
      atomicAdd(&sum[o], s[r]);
      atomicAdd(&sumsq[o], q[r]);
    }
  }
}

template <int NCH>
__global__ void edge_apply_kernel(const _Float16* __restrict__ X, int RS,
                                  const int* __restrict__ idx,
                                  const _Float16* __restrict__ Wp,
                                  const float* __restrict__ scale,
                                  const float* __restrict__ shift,
                                  float* __restrict__ feat, int Co, int TPW) {
  const int twoC = NCH * 32;
  int wave = threadIdx.x >> 5, lane = threadIdx.x & 31;
  int l15 = lane & 15, hi = lane >> 4;
  int b = blockIdx.z, obase = blockIdx.x * 16;
  const _Float16* Xb = X + (long)b * NN * RS;
  const int* idxb = idx + (long)b * NN * KK;
  const int CT = NN * KK / 16;

  v16h af[NCH];
  const _Float16* wrow = Wp + (long)(obase + l15) * twoC;
#pragma unroll
  for (int k = 0; k < NCH; ++k) af[k] = load_afrag(wrow, k * 32, hi);

  float sc[8], sh[8];
#pragma unroll
  for (int r = 0; r < 8; ++r) {
    int o = obase + r + (hi << 3);
    sc[r] = scale[o]; sh[r] = shift[o];
  }

  for (int it = 0; it < TPW; ++it) {
    int t = (blockIdx.y * 8 + wave) * TPW + it;
    if (t >= CT) break;
    int j = t * 16 + l15;
    int n = j / KK;
    int m = idxb[j];
    const _Float16* Xn = Xb + (long)n * RS;
    const _Float16* Xm = Xb + (long)m * RS;
    v8f acc = {};
#pragma unroll
    for (int k = 0; k < NCH; ++k)
      acc = wmma16(af[k], edge_bfrag<NCH>(Xn, Xm, k, hi), acc);
#pragma unroll
    for (int r = 0; r < 8; ++r) {
      int o = obase + r + (hi << 3);
      float y = fmaxf(acc[r] * sc[r] + sh[r], 0.0f);
      // relu output >= 0: int compare of bits == float compare; feat pre-zeroed
      atomicMax((int*)&feat[((long)b * Co + o) * NN + n], __float_as_int(y));
    }
  }
}

// feat (B,Co,N) f32 -> cat (B,N,448) f16 at channel offset
__global__ void feat_to_cat_kernel(const float* __restrict__ feat, _Float16* __restrict__ cat,
                                   int Co, int chOff, long total) {
  long t = (long)blockIdx.x * blockDim.x + threadIdx.x;
  if (t >= total) return;
  long b = t / ((long)Co * NN);
  long r = t - b * (long)Co * NN;
  int c = (int)(r / NN), n = (int)(r - (long)c * NN);
  cat[((long)b * NN + n) * 448 + chOff + c] = (_Float16)feat[t];
}

// ---------------------------------------------------------------------------
// Final 1x1 conv (NCH = 14): y[o,(b,n)] = sum_c Wf[o,c] * cat[(b,n),c]
// ---------------------------------------------------------------------------
template <int NCH>
__global__ void mlp_stats_kernel(const _Float16* __restrict__ cat,
                                 const _Float16* __restrict__ Wp,
                                 float* __restrict__ sum, float* __restrict__ sumsq,
                                 int TPW) {
  const int twoC = NCH * 32;
  int wave = threadIdx.x >> 5, lane = threadIdx.x & 31;
  int l15 = lane & 15, hi = lane >> 4;
  int obase = blockIdx.x * 16;
  const int CT = BB * NN / 16;

  v16h af[NCH];
  const _Float16* wrow = Wp + (long)(obase + l15) * twoC;
#pragma unroll
  for (int k = 0; k < NCH; ++k) af[k] = load_afrag(wrow, k * 32, hi);

  float s[8] = {0, 0, 0, 0, 0, 0, 0, 0}, q[8] = {0, 0, 0, 0, 0, 0, 0, 0};
  for (int it = 0; it < TPW; ++it) {
    int t = (blockIdx.y * 8 + wave) * TPW + it;
    if (t >= CT) break;
    int j = t * 16 + l15;                     // global column = b*N + n
    const _Float16* row = cat + (long)j * twoC;
    v8f acc = {};
#pragma unroll
    for (int k = 0; k < NCH; ++k)
      acc = wmma16(af[k], load_bfrag(row, k * 32, hi), acc);
#pragma unroll
    for (int r = 0; r < 8; ++r) { float y = acc[r]; s[r] += y; q[r] += y * y; }
  }
#pragma unroll
  for (int r = 0; r < 8; ++r)
#pragma unroll
    for (int off = 8; off > 0; off >>= 1) {
      s[r] += __shfl_xor(s[r], off, 16);
      q[r] += __shfl_xor(q[r], off, 16);
    }
  if (l15 == 0) {
#pragma unroll
    for (int r = 0; r < 8; ++r) {
      int o = obase + r + (hi << 3);
      atomicAdd(&sum[o], s[r]);
      atomicAdd(&sumsq[o], q[r]);
    }
  }
}

template <int NCH>
__global__ void mlp_apply_kernel(const _Float16* __restrict__ cat,
                                 const _Float16* __restrict__ Wp,
                                 const float* __restrict__ scale,
                                 const float* __restrict__ shift,
                                 float* __restrict__ out, int TPW) {
  const int twoC = NCH * 32;
  int wave = threadIdx.x >> 5, lane = threadIdx.x & 31;
  int l15 = lane & 15, hi = lane >> 4;
  int obase = blockIdx.x * 16;
  const int CT = BB * NN / 16;

  v16h af[NCH];
  const _Float16* wrow = Wp + (long)(obase + l15) * twoC;
#pragma unroll
  for (int k = 0; k < NCH; ++k) af[k] = load_afrag(wrow, k * 32, hi);

  float sc[8], sh[8];
#pragma unroll
  for (int r = 0; r < 8; ++r) {
    int o = obase + r + (hi << 3);
    sc[r] = scale[o]; sh[r] = shift[o];
  }

  for (int it = 0; it < TPW; ++it) {
    int t = (blockIdx.y * 8 + wave) * TPW + it;
    if (t >= CT) break;
    int j = t * 16 + l15;
    int b = j / NN, n = j - b * NN;
    const _Float16* row = cat + (long)j * twoC;
    v8f acc = {};
#pragma unroll
    for (int k = 0; k < NCH; ++k)
      acc = wmma16(af[k], load_bfrag(row, k * 32, hi), acc);
#pragma unroll
    for (int r = 0; r < 8; ++r) {
      int o = obase + r + (hi << 3);
      out[((long)b * 1024 + o) * NN + n] = fmaxf(acc[r] * sc[r] + sh[r], 0.0f);
    }
  }
}

// BN: scale = g * rsqrt(var+eps); shift = b - mean*scale
__global__ void bnparam_kernel(const float* __restrict__ sum, const float* __restrict__ sumsq,
                               const float* __restrict__ g, const float* __restrict__ bta,
                               float* __restrict__ scale, float* __restrict__ shift,
                               int co, float invCount) {
  int o = blockIdx.x * blockDim.x + threadIdx.x;
  if (o >= co) return;
  float m   = sum[o] * invCount;
  float var = sumsq[o] * invCount - m * m;
  float rs  = rsqrtf(var + 1e-5f);
  float sc  = g[o] * rs;
  scale[o] = sc;
  shift[o] = bta[o] - m * sc;
}

// ---------------------------------------------------------------------------
// Host orchestration
// ---------------------------------------------------------------------------
template <int NCH>
static void run_layer(const _Float16* X, int RS,
                      const _Float16* Wp,
                      const float* g, const float* bb, int Co, int chOff,
                      float* dist, int* idx, float* sq, float* feat,
                      float* sum, float* sumsq, float* scale, float* shift,
                      _Float16* cat, hipStream_t stream) {
  const int Cin = NCH * 16;
  sqnorm_kernel<<<BB * NN / 256, 256, 0, stream>>>(X, RS, sq, Cin);
  dim3 gg(NN / 128, NN / 16, BB);
  gram_dist_kernel<<<gg, 256, 0, stream>>>(X, RS, sq, dist, Cin);
  topk_kernel<<<BB * NN * 32 / 256, 256, 0, stream>>>(dist, idx);

  hipMemsetAsync(sum, 0, Co * 4, stream);
  hipMemsetAsync(sumsq, 0, Co * 4, stream);
  const int TPW = 32;
  int CT = NN * KK / 16;
  int strips = (CT + 8 * TPW - 1) / (8 * TPW);
  dim3 gs(Co / 16, strips, BB);
  edge_stats_kernel<NCH><<<gs, 256, 0, stream>>>(X, RS, idx, Wp, sum, sumsq, TPW);
  bnparam_kernel<<<(Co + 255) / 256, 256, 0, stream>>>(sum, sumsq, g, bb, scale, shift,
                                                       Co, 1.0f / (float)(BB * NN * KK));
  hipMemsetAsync(feat, 0, (size_t)BB * Co * NN * 4, stream);
  edge_apply_kernel<NCH><<<gs, 256, 0, stream>>>(X, RS, idx, Wp, scale, shift, feat, Co, TPW);
  long total = (long)BB * Co * NN;
  feat_to_cat_kernel<<<(unsigned)((total + 255) / 256), 256, 0, stream>>>(feat, cat, Co,
                                                                          chOff, total);
}

extern "C" void kernel_launch(void* const* d_in, const int* in_sizes, int n_in,
                              void* d_out, int out_size, void* d_ws, size_t ws_size,
                              hipStream_t stream) {
  const float* x  = (const float*)d_in[0];
  const float* W1 = (const float*)d_in[1];
  const float* g1 = (const float*)d_in[2];
  const float* b1 = (const float*)d_in[3];
  const float* W2 = (const float*)d_in[4];
  const float* g2 = (const float*)d_in[5];
  const float* b2 = (const float*)d_in[6];
  const float* W3 = (const float*)d_in[7];
  const float* g3 = (const float*)d_in[8];
  const float* b3 = (const float*)d_in[9];
  const float* Wf = (const float*)d_in[10];
  const float* gf = (const float*)d_in[11];
  const float* bf = (const float*)d_in[12];
  float* out = (float*)d_out;

  char* ws = (char*)d_ws;
  size_t off = 0;
  auto alloc = [&](size_t bytes) -> void* {
    off = (off + 255) & ~(size_t)255;
    void* p = ws + off;
    off += bytes;
    return p;
  };
  _Float16* xpad = (_Float16*)alloc((size_t)BB * NN * 32 * 2);   // (B,N,32)
  _Float16* cat  = (_Float16*)alloc((size_t)BB * NN * 448 * 2);  // (B,N,448)
  _Float16* w1p  = (_Float16*)alloc((size_t)64 * 64 * 2);
  _Float16* w2p  = (_Float16*)alloc((size_t)128 * 128 * 2);
  _Float16* w3p  = (_Float16*)alloc((size_t)256 * 256 * 2);
  _Float16* wfp  = (_Float16*)alloc((size_t)1024 * 448 * 2);
  float* dist    = (float*)alloc((size_t)BB * NN * NN * 4);
  int*   idx     = (int*)alloc((size_t)BB * NN * KK * 4);
  float* sq      = (float*)alloc((size_t)BB * NN * 4);
  float* feat    = (float*)alloc((size_t)BB * 256 * NN * 4);
  float* sum     = (float*)alloc(1024 * 4);
  float* sumsq   = (float*)alloc(1024 * 4);
  float* scale   = (float*)alloc(1024 * 4);
  float* shift   = (float*)alloc(1024 * 4);

  prep_x_kernel<<<BB * NN / 256, 256, 0, stream>>>(x, xpad);
  prep_w_kernel<<<(64 * 64 + 255) / 256,   256, 0, stream>>>(W1, w1p, 64, 3, 32, 64);
  prep_w_kernel<<<(128 * 128 + 255) / 256, 256, 0, stream>>>(W2, w2p, 128, 64, 64, 128);
  prep_w_kernel<<<(256 * 256 + 255) / 256, 256, 0, stream>>>(W3, w3p, 256, 128, 128, 256);
  w_flat_kernel<<<(1024 * 448 + 255) / 256, 256, 0, stream>>>(Wf, wfp, (long)1024 * 448);

  // layer 1: X = xpad (row stride 32, Cin 32 padded), layer 2/3: slices of cat
  run_layer<2>(xpad,      32,  w1p, g1, b1, 64,  0,
               dist, idx, sq, feat, sum, sumsq, scale, shift, cat, stream);
  run_layer<4>(cat,       448, w2p, g2, b2, 128, 64,
               dist, idx, sq, feat, sum, sumsq, scale, shift, cat, stream);
  run_layer<8>(cat + 64,  448, w3p, g3, b3, 256, 192,
               dist, idx, sq, feat, sum, sumsq, scale, shift, cat, stream);

  // final conv
  hipMemsetAsync(sum, 0, 1024 * 4, stream);
  hipMemsetAsync(sumsq, 0, 1024 * 4, stream);
  const int TPW2 = 8;
  int CT2 = BB * NN / 16;
  int strips2 = (CT2 + 8 * TPW2 - 1) / (8 * TPW2);
  dim3 gm(1024 / 16, strips2, 1);
  mlp_stats_kernel<14><<<gm, 256, 0, stream>>>(cat, wfp, sum, sumsq, TPW2);
  bnparam_kernel<<<(1024 + 255) / 256, 256, 0, stream>>>(sum, sumsq, gf, bf, scale, shift,
                                                         1024, 1.0f / (float)(BB * NN));
  mlp_apply_kernel<14><<<gm, 256, 0, stream>>>(cat, wfp, scale, shift, out, TPW2);
}
